// Decoder_83150566850667
// MI455X (gfx1250) — matrix operand order repaired and checked by reading, not compiled, not run on hardware
//
#include <hip/hip_runtime.h>
#include <hip/hip_bf16.h>

// ---- problem constants --------------------------------------------------
#define BDIM   32      // batch
#define TLEN   40      // caption length (steps = TLEN-1 = 39)
#define TSTEPS 39
#define VDIM   8000    // vocab
#define EDIM   50      // embedding dim
#define EPAD   64      // padded embedding dim (multiple of 32 for WMMA K)
#define HDIM   2048    // hidden
#define KSLICE (HDIM / 4)   // per-wave K slice (512)

typedef __attribute__((ext_vector_type(16))) __bf16 v16bf;
typedef __attribute__((ext_vector_type(8)))  float  v8f;

union AFrag { uint4 q[2]; v16bf v; };

// A-matrix fragment (16x32 bf16, row-major source [M x ld]).
// lane 0-15: row m=l15, K = k0+lh*8 .. +7  and  k0+16+lh*8 .. +7
static __device__ __forceinline__
v16bf load_a(const __bf16* __restrict__ X, int ld, int mbase, int k0, int l15, int lh) {
  const __bf16* p = X + (size_t)(mbase + l15) * ld + k0 + lh * 8;
  AFrag u;
  u.q[0] = *(const uint4*)(p);        // K = k0+lh*8 .. +7
  u.q[1] = *(const uint4*)(p + 16);   // K = k0+16+lh*8 .. +7
  return u.v;
}

// B-matrix fragment (32x16 bf16). B[k,n] = W[n,k] with W row-major [N x ld],
// so a lane's column is two 16-byte slices of a weight row: K = k0+lh*8 .. +7 and k0+16+lh*8 .. +7.
static __device__ __forceinline__
v16bf load_b(const __bf16* __restrict__ W, int ld, int nbase, int k0, int l15, int lh) {
  const __bf16* p = W + (size_t)(nbase + l15) * ld + k0 + lh * 8;
  AFrag u;
  u.q[0] = *(const uint4*)(p);
  u.q[1] = *(const uint4*)(p + 16);
  return u.v;
}

static __device__ __forceinline__
v8f wmma_bf(v16bf a, v16bf b, v8f c) {
  // (neg_a, A, neg_b, B, c_mod, C, reuse_a, reuse_b)
  return __builtin_amdgcn_wmma_f32_16x16x32_bf16(false, a, false, b, (short)0, c, false, false);
}

static __device__ __forceinline__ float sigm(float x) { return 1.0f / (1.0f + __expf(-x)); }

// ---- prep kernels -------------------------------------------------------
__global__ void k_f2bf(const float* __restrict__ s, __bf16* __restrict__ d, int n) {
  int i = blockIdx.x * blockDim.x + threadIdx.x;
  if (i < n) d[i] = (__bf16)s[i];
}

__global__ void k_pad_wih(const float* __restrict__ s, __bf16* __restrict__ d) { // [4H,50]->[4H,64]
  int i = blockIdx.x * blockDim.x + threadIdx.x;
  if (i >= 4 * HDIM * EPAD) return;
  int r = i >> 6, c = i & 63;
  d[i] = (c < EDIM) ? (__bf16)s[r * EDIM + c] : (__bf16)0.0f;
}

__global__ void k_bias_sum(const float* __restrict__ a, const float* __restrict__ b,
                           float* __restrict__ d, int n) {
  int i = blockIdx.x * blockDim.x + threadIdx.x;
  if (i < n) d[i] = a[i] + b[i];
}

__global__ void k_embed(const int* __restrict__ cap, const float* __restrict__ emb,
                        __bf16* __restrict__ x) {  // x [39][32][64]
  int i = blockIdx.x * blockDim.x + threadIdx.x;
  if (i >= TSTEPS * BDIM * EPAD) return;
  int e = i & 63, b = (i >> 6) & 31, t = i >> 11;
  int c = cap[b * TLEN + t];
  x[i] = (e < EDIM) ? (__bf16)emb[c * EDIM + e] : (__bf16)0.0f;
}

__global__ void k_init(const float* __restrict__ f, float* __restrict__ c32,
                       float* __restrict__ hg32, __bf16* __restrict__ hbf,
                       __bf16* __restrict__ hgbf) {
  int i = blockIdx.x * blockDim.x + threadIdx.x;
  if (i >= BDIM * HDIM) return;
  float v = f[i];
  c32[i] = v; hg32[i] = v;
  hbf[i] = (__bf16)v; hgbf[i] = (__bf16)v;
}

// ---- LSTM cell: split-K gates GEMM + LDS reduce + fused state update ----
// grid = H/16 blocks, 128 threads. Block owns 16 columns; each wave does a
// K/4 slice of the h GEMM (wave 0 also does all of the x GEMM); partials reduced via LDS; 128 threads then do the
// elementwise LSTM update (4 outputs each).
__global__ __launch_bounds__(128) void k_lstm(
    const __bf16* __restrict__ xbf,    // [32,64]
    const __bf16* __restrict__ hbf_in, // [32,H]
    const __bf16* __restrict__ WihP,   // [4H,64]
    const __bf16* __restrict__ Whh,    // [4H,H]
    const float*  __restrict__ bias,   // [4H]  (bih+bhh)
    float*        __restrict__ c32,    // [32,H] in/out (owner-only)
    __bf16*       __restrict__ hbf_out,
    __bf16*       __restrict__ cbf_out) {
  const int lane = threadIdx.x & 31, wave = threadIdx.x >> 5;
  const int l15 = lane & 15, lh = lane >> 4;
  const int n0 = blockIdx.x * 16;

  __shared__ float red[4][2][4][16][16];   // [wave][mt][gate][m16][n16] = 32 KB

  v8f acc[2][4];
#pragma unroll
  for (int mt = 0; mt < 2; ++mt)
#pragma unroll
    for (int g = 0; g < 4; ++g) acc[mt][g] = (v8f)0.0f;

  // x @ Wih^T (K = 64 padded) — tiny, wave 0 only
  if (wave == 0) {
#pragma unroll
    for (int k0 = 0; k0 < EPAD; k0 += 32) {
      v16bf a0 = load_a(xbf, EPAD, 0, k0, l15, lh);
      v16bf a1 = load_a(xbf, EPAD, 16, k0, l15, lh);
#pragma unroll
      for (int g = 0; g < 4; ++g) {
        v16bf b = load_b(WihP, EPAD, g * HDIM + n0, k0, l15, lh);
        acc[0][g] = wmma_bf(a0, b, acc[0][g]);
        acc[1][g] = wmma_bf(a1, b, acc[1][g]);
      }
    }
  }
  // h @ Whh^T (K = 2048, this wave's 512-slice)
  const int kb = wave * KSLICE;
  for (int kk = 0; kk < KSLICE; kk += 32) {
    const int k0 = kb + kk;
    __builtin_prefetch(Whh + (size_t)(n0 + l15) * HDIM + k0 + 256, 0, 3);
    v16bf a0 = load_a(hbf_in, HDIM, 0, k0, l15, lh);
    v16bf a1 = load_a(hbf_in, HDIM, 16, k0, l15, lh);
#pragma unroll
    for (int g = 0; g < 4; ++g) {
      v16bf b = load_b(Whh, HDIM, g * HDIM + n0, k0, l15, lh);
      acc[0][g] = wmma_bf(a0, b, acc[0][g]);
      acc[1][g] = wmma_bf(a1, b, acc[1][g]);
    }
  }

  // dump partials to LDS:  D element (v, lane) is (m = lh*8+v, n = l15)
#pragma unroll
  for (int mt = 0; mt < 2; ++mt)
#pragma unroll
    for (int g = 0; g < 4; ++g)
#pragma unroll
      for (int v = 0; v < 8; ++v)
        red[wave][mt][g][lh * 8 + v][l15] = acc[mt][g][v];
  __syncthreads();

  // reduce over 4 waves + fused LSTM nonlinearity; 4 outputs per thread
  const int e0 = threadIdx.x * 4;
#pragma unroll
  for (int e = e0; e < e0 + 4; ++e) {
    const int m = e >> 4, n16 = e & 15;
    const int mt = m >> 4, m16 = m & 15;
    const int j = n0 + n16;
    float gi = 0.f, gf = 0.f, gg = 0.f, go = 0.f;
#pragma unroll
    for (int w = 0; w < 4; ++w) {
      gi += red[w][mt][0][m16][n16];
      gf += red[w][mt][1][m16][n16];
      gg += red[w][mt][2][m16][n16];
      go += red[w][mt][3][m16][n16];
    }
    float iv = sigm(gi + bias[j]);
    float fv = sigm(gf + bias[HDIM + j]);
    float gv = tanhf(gg + bias[2 * HDIM + j]);
    float ov = sigm(go + bias[3 * HDIM + j]);
    float cold = c32[m * HDIM + j];
    float cnew = fv * cold + iv * gv;
    float hnew = ov * tanhf(cnew);
    c32[m * HDIM + j] = cnew;
    cbf_out[m * HDIM + j] = (__bf16)cnew;
    hbf_out[m * HDIM + j] = (__bf16)hnew;
  }
}

// ---- GRU cell (input = LSTM cell state c), same split-K scheme ----------
__global__ __launch_bounds__(128) void k_gru(
    const __bf16* __restrict__ cbf,      // [32,H]
    const __bf16* __restrict__ hgbf_in,  // [32,H]
    const __bf16* __restrict__ gWih,     // [3H,H]
    const __bf16* __restrict__ gWhh,     // [3H,H]
    const float*  __restrict__ gbih,     // [3H]
    const float*  __restrict__ gbhh,     // [3H]
    float*        __restrict__ hg32,     // [32,H] in/out (owner-only)
    __bf16*       __restrict__ hgbf_out) {
  const int lane = threadIdx.x & 31, wave = threadIdx.x >> 5;
  const int l15 = lane & 15, lh = lane >> 4;
  const int n0 = blockIdx.x * 16;

  __shared__ float red[4][2][6][16][16];   // [wave][mt][chunk(gi:0-2, gh:3-5)][m16][n16] = 48 KB

  v8f ai[2][3], ah[2][3];
#pragma unroll
  for (int mt = 0; mt < 2; ++mt)
#pragma unroll
    for (int ch = 0; ch < 3; ++ch) { ai[mt][ch] = (v8f)0.0f; ah[mt][ch] = (v8f)0.0f; }

  const int kb = wave * KSLICE;
  // gi = c @ gWih^T
  for (int kk = 0; kk < KSLICE; kk += 32) {
    const int k0 = kb + kk;
    __builtin_prefetch(gWih + (size_t)(n0 + l15) * HDIM + k0 + 256, 0, 3);
    v16bf a0 = load_a(cbf, HDIM, 0, k0, l15, lh);
    v16bf a1 = load_a(cbf, HDIM, 16, k0, l15, lh);
#pragma unroll
    for (int ch = 0; ch < 3; ++ch) {
      v16bf b = load_b(gWih, HDIM, ch * HDIM + n0, k0, l15, lh);
      ai[0][ch] = wmma_bf(a0, b, ai[0][ch]);
      ai[1][ch] = wmma_bf(a1, b, ai[1][ch]);
    }
  }
  // gh = hg @ gWhh^T
  for (int kk = 0; kk < KSLICE; kk += 32) {
    const int k0 = kb + kk;
    __builtin_prefetch(gWhh + (size_t)(n0 + l15) * HDIM + k0 + 256, 0, 3);
    v16bf a0 = load_a(hgbf_in, HDIM, 0, k0, l15, lh);
    v16bf a1 = load_a(hgbf_in, HDIM, 16, k0, l15, lh);
#pragma unroll
    for (int ch = 0; ch < 3; ++ch) {
      v16bf b = load_b(gWhh, HDIM, ch * HDIM + n0, k0, l15, lh);
      ah[0][ch] = wmma_bf(a0, b, ah[0][ch]);
      ah[1][ch] = wmma_bf(a1, b, ah[1][ch]);
    }
  }

#pragma unroll
  for (int mt = 0; mt < 2; ++mt)
#pragma unroll
    for (int ch = 0; ch < 3; ++ch)
#pragma unroll
      for (int v = 0; v < 8; ++v) {
        red[wave][mt][ch][lh * 8 + v][l15]     = ai[mt][ch][v];
        red[wave][mt][ch + 3][lh * 8 + v][l15] = ah[mt][ch][v];
      }
  __syncthreads();

  const int e0 = threadIdx.x * 4;
#pragma unroll
  for (int e = e0; e < e0 + 4; ++e) {
    const int m = e >> 4, n16 = e & 15;
    const int mt = m >> 4, m16 = m & 15;
    const int j = n0 + n16;
    float ri = 0.f, zi = 0.f, ni = 0.f, rh = 0.f, zh = 0.f, nh = 0.f;
#pragma unroll
    for (int w = 0; w < 4; ++w) {
      ri += red[w][mt][0][m16][n16];  zi += red[w][mt][1][m16][n16];
      ni += red[w][mt][2][m16][n16];  rh += red[w][mt][3][m16][n16];
      zh += red[w][mt][4][m16][n16];  nh += red[w][mt][5][m16][n16];
    }
    float r = sigm(ri + gbih[j] + rh + gbhh[j]);
    float z = sigm(zi + gbih[HDIM + j] + zh + gbhh[HDIM + j]);
    float nn = tanhf(ni + gbih[2 * HDIM + j] + r * (nh + gbhh[2 * HDIM + j]));
    float hold = hg32[m * HDIM + j];
    float hnew = (1.0f - z) * nn + z * hold;
    hg32[m * HDIM + j] = hnew;
    hgbf_out[m * HDIM + j] = (__bf16)hnew;
  }
}

// ---- vocab projection: out[b, t, :] = hg @ lin_W^T + lin_b --------------
__global__ __launch_bounds__(128) void k_lin(
    const __bf16* __restrict__ hgbf,  // [32,H]
    const __bf16* __restrict__ Wl,    // [V,H]
    const float*  __restrict__ bl,    // [V]
    float*        __restrict__ out,   // [B, 39, V]
    int t) {
  const int lane = threadIdx.x & 31, wave = threadIdx.x >> 5;
  const int l15 = lane & 15, lh = lane >> 4;
  const int n0 = blockIdx.x * 16;   // 500 blocks * 16 = 8000

  __shared__ float red[4][2][16][16];   // 8 KB

  v8f acc[2];
  acc[0] = (v8f)0.0f; acc[1] = (v8f)0.0f;
  const int kb = wave * KSLICE;
  for (int kk = 0; kk < KSLICE; kk += 32) {
    const int k0 = kb + kk;
    __builtin_prefetch(Wl + (size_t)(n0 + l15) * HDIM + k0 + 256, 0, 3);
    v16bf a0 = load_a(hgbf, HDIM, 0, k0, l15, lh);
    v16bf a1 = load_a(hgbf, HDIM, 16, k0, l15, lh);
    v16bf b = load_b(Wl, HDIM, n0, k0, l15, lh);
    acc[0] = wmma_bf(a0, b, acc[0]);
    acc[1] = wmma_bf(a1, b, acc[1]);
  }
#pragma unroll
  for (int mt = 0; mt < 2; ++mt)
#pragma unroll
    for (int v = 0; v < 8; ++v)
      red[wave][mt][lh * 8 + v][l15] = acc[mt][v];
  __syncthreads();

  const int e0 = threadIdx.x * 4;
#pragma unroll
  for (int e = e0; e < e0 + 4; ++e) {
    const int m = e >> 4, n16 = e & 15;
    const int mt = m >> 4, m16 = m & 15;
    const int j = n0 + n16;
    float s = bl[j];
#pragma unroll
    for (int w = 0; w < 4; ++w) s += red[w][mt][m16][n16];
    out[(size_t)m * (TSTEPS * VDIM) + (size_t)t * VDIM + j] = s;
  }
}

// ---- host launcher ------------------------------------------------------
extern "C" void kernel_launch(void* const* d_in, const int* in_sizes, int n_in,
                              void* d_out, int out_size, void* d_ws, size_t ws_size,
                              hipStream_t stream) {
  const float* features = (const float*)d_in[0];
  const int*   captions = (const int*)d_in[1];
  const float* emb      = (const float*)d_in[2];
  const float* lstm_Wih = (const float*)d_in[3];
  const float* lstm_bih = (const float*)d_in[4];
  const float* lstm_Whh = (const float*)d_in[5];
  const float* lstm_bhh = (const float*)d_in[6];
  const float* gru_Wih  = (const float*)d_in[7];
  const float* gru_bih  = (const float*)d_in[8];
  const float* gru_Whh  = (const float*)d_in[9];
  const float* gru_bhh  = (const float*)d_in[10];
  const float* lin_W    = (const float*)d_in[11];
  const float* lin_b    = (const float*)d_in[12];
  float* out = (float*)d_out;

  // workspace carve-out (256B aligned slices)
  size_t off = 0;
  auto take = [&](size_t bytes) -> void* {
    void* p = (char*)d_ws + off;
    off += (bytes + 255) & ~(size_t)255;
    return p;
  };
  __bf16* wihp  = (__bf16*)take((size_t)4 * HDIM * EPAD * 2);
  __bf16* whh   = (__bf16*)take((size_t)4 * HDIM * HDIM * 2);
  __bf16* gwih  = (__bf16*)take((size_t)3 * HDIM * HDIM * 2);
  __bf16* gwhh  = (__bf16*)take((size_t)3 * HDIM * HDIM * 2);
  __bf16* wlin  = (__bf16*)take((size_t)VDIM * HDIM * 2);
  __bf16* xbf   = (__bf16*)take((size_t)TSTEPS * BDIM * EPAD * 2);
  float*  lbias = (float*)take((size_t)4 * HDIM * 4);
  float*  c32   = (float*)take((size_t)BDIM * HDIM * 4);
  float*  hg32  = (float*)take((size_t)BDIM * HDIM * 4);
  __bf16* cbf   = (__bf16*)take((size_t)BDIM * HDIM * 2);
  __bf16* hbf[2], *hgbf[2];
  hbf[0]  = (__bf16*)take((size_t)BDIM * HDIM * 2);
  hbf[1]  = (__bf16*)take((size_t)BDIM * HDIM * 2);
  hgbf[0] = (__bf16*)take((size_t)BDIM * HDIM * 2);
  hgbf[1] = (__bf16*)take((size_t)BDIM * HDIM * 2);

  auto cdiv = [](int a, int b) { return (a + b - 1) / b; };
  const int THR = 256;

  // ---- prep (runs every call; deterministic) ----
  {
    int n = 4 * HDIM * HDIM;
    k_f2bf<<<cdiv(n, THR), THR, 0, stream>>>(lstm_Whh, whh, n);
    n = 3 * HDIM * HDIM;
    k_f2bf<<<cdiv(n, THR), THR, 0, stream>>>(gru_Wih, gwih, n);
    k_f2bf<<<cdiv(n, THR), THR, 0, stream>>>(gru_Whh, gwhh, n);
    n = VDIM * HDIM;
    k_f2bf<<<cdiv(n, THR), THR, 0, stream>>>(lin_W, wlin, n);
    n = 4 * HDIM * EPAD;
    k_pad_wih<<<cdiv(n, THR), THR, 0, stream>>>(lstm_Wih, wihp);
    n = 4 * HDIM;
    k_bias_sum<<<cdiv(n, THR), THR, 0, stream>>>(lstm_bih, lstm_bhh, lbias, n);
    n = TSTEPS * BDIM * EPAD;
    k_embed<<<cdiv(n, THR), THR, 0, stream>>>(captions, emb, xbf);
    n = BDIM * HDIM;
    k_init<<<cdiv(n, THR), THR, 0, stream>>>(features, c32, hg32, hbf[0], hgbf[0]);
  }

  // ---- 39 sequential decoder steps ----
  for (int t = 0; t < TSTEPS; ++t) {
    int in = t & 1, outb = (t + 1) & 1;
    k_lstm<<<HDIM / 16, 128, 0, stream>>>(xbf + (size_t)t * BDIM * EPAD, hbf[in],
                                          wihp, whh, lbias, c32, hbf[outb], cbf);
    k_gru<<<HDIM / 16, 128, 0, stream>>>(cbf, hgbf[in], gwih, gwhh, gru_bih, gru_bhh,
                                         hg32, hgbf[outb]);
    k_lin<<<VDIM / 16, 128, 0, stream>>>(hgbf[outb], wlin, lin_b, out, t);
  }
}
